// graphSAGELayer_54941221650896
// MI455X (gfx1250) — compile-verified, matrix operand order checked by reading, not run on hardware
//
#include <hip/hip_runtime.h>
#include <hip/hip_bf16.h>
#include <stdint.h>

// Problem constants (from reference)
#define N_NODES 50000
#define KNEI    20
#define FDIM    256
#define DDIM    128
#define KC      256   // fused K for [x_t | h_t] @ [W_ih | W_hh]^T

typedef __attribute__((ext_vector_type(16))) __bf16 v16bf;
typedef __attribute__((ext_vector_type(8)))  float  v8f;

union FragBF { v16bf v; uint4 q[2]; };

// Native hardware conversion (single v_cvt op) instead of software RNE.
static __device__ __forceinline__ unsigned short f2bf(float f) {
  __bf16 b = (__bf16)f;
  return __builtin_bit_cast(unsigned short, b);
}
static __device__ __forceinline__ unsigned int pack2(float a, float b) {
  return (unsigned int)f2bf(a) | ((unsigned int)f2bf(b) << 16);
}
static __device__ __forceinline__ float sigmoidf_(float x) {
  return 1.0f / (1.0f + expf(-x));
}

// ---------------------------------------------------------------------------
// Kernel 0: weight preprocessing.
//  WpBt[n][k] = bf16(W_past[k][n])            (128 x 256, n-major "Bt" layout)
//  WBt [n][k] = bf16(W[k][n])                 (128 x 256)
//  WcBt[n][k] = bf16(k<128 ? W_ih[n][k] : W_hh[n][k-128])   (512 x 256)
//  bsum[n]    = b_ih[n] + b_hh[n]
// ---------------------------------------------------------------------------
__global__ void prep_kernel(const float* __restrict__ Wpast, const float* __restrict__ W,
                            const float* __restrict__ Wih,   const float* __restrict__ Whh,
                            const float* __restrict__ bih,   const float* __restrict__ bhh,
                            unsigned short* __restrict__ WpBt, unsigned short* __restrict__ WBt,
                            unsigned short* __restrict__ WcBt, float* __restrict__ bsum) {
  int i = blockIdx.x * blockDim.x + threadIdx.x;
  if (i < 128 * 256) {
    int n = i >> 8, k = i & 255;
    WpBt[i] = f2bf(Wpast[k * DDIM + n]);
    WBt[i]  = f2bf(W[k * DDIM + n]);
  }
  if (i < 512 * 256) {
    int n = i >> 8, k = i & 255;
    WcBt[i] = f2bf(k < 128 ? Wih[n * DDIM + k] : Whh[n * DDIM + (k - 128)]);
  }
  if (i < 512) bsum[i] = bih[i] + bhh[i];
}

// ---------------------------------------------------------------------------
// Kernel 0b: h (N x 256 f32) -> hB (bf16), bandwidth-bound one-shot convert.
// Keeps all f32->bf16 VALU out of the GEMM inner loops.
// ---------------------------------------------------------------------------
__global__ void hcvt_kernel(const float* __restrict__ h,
                            unsigned short* __restrict__ hB) {
  size_t i = ((size_t)blockIdx.x * blockDim.x + threadIdx.x) * 8;
  if (i < (size_t)N_NODES * FDIM) {
    const float4* p = (const float4*)(h + i);
    float4 a = p[0], b = p[1];
    *(uint4*)(hB + i) = make_uint4(pack2(a.x, a.y), pack2(a.z, a.w),
                                   pack2(b.x, b.y), pack2(b.z, b.w));
  }
}

// ---------------------------------------------------------------------------
// Kernel 1: hp = h @ W_past   (N x 256)@(256 x 128) -> bf16 hp (N x 128)
// WG = 256 threads (8 waves), 64 rows; wave w owns output cols [16w,16w+16).
// Pure load->wmma inner loop (A already bf16).
// ---------------------------------------------------------------------------
__global__ void hp_kernel(const unsigned short* __restrict__ hB,
                          unsigned short* __restrict__ hpB,
                          const unsigned short* __restrict__ WpBt) {
  extern __shared__ __align__(16) unsigned short smem[];   // 64 KB: WpBt
  const int tid = threadIdx.x;
  {
    uint4* d = (uint4*)smem;
    const uint4* s = (const uint4*)WpBt;
    for (int i = tid; i < (128 * 256) / 8; i += blockDim.x) d[i] = s[i];
  }
  __syncthreads();
  const int wave = tid >> 5, lane = tid & 31;
  const int ln = lane & 15, hi = lane >> 4;
  const int row0 = blockIdx.x * 64;

  for (int rt = 0; rt < 4; ++rt) {
    int row = row0 + rt * 16 + ln;
    int rowc = row < N_NODES ? row : N_NODES - 1;
    const unsigned short* arow = hB + (size_t)rowc * FDIM;
    v8f acc = {};
#pragma unroll
    for (int kc = 0; kc < 8; ++kc) {
      const int k0 = kc * 32;
      FragBF a;
      a.q[0] = *(const uint4*)(arow + k0 + 8 * hi);
      a.q[1] = *(const uint4*)(arow + k0 + 16 + 8 * hi);
      const unsigned short* brow = smem + (wave * 16 + ln) * KC;
      FragBF b;
      b.q[0] = *(const uint4*)(brow + k0 + 8 * hi);
      b.q[1] = *(const uint4*)(brow + k0 + 16 + 8 * hi);
      acc = __builtin_amdgcn_wmma_f32_16x16x32_bf16(false, a.v, false, b.v,
                                                    (short)0, acc, false, false);
    }
    const int col = wave * 16 + ln;
#pragma unroll
    for (int v = 0; v < 8; ++v) {
      int r = row0 + rt * 16 + v + 8 * hi;   // C layout: M = v + 8*hi, N = ln
      if (r < N_NODES) hpB[(size_t)r * DDIM + col] = f2bf(acc[v]);
    }
  }
}

// ---------------------------------------------------------------------------
// Kernel 2: fused 20-step LSTM over gathered neighbor sequences.
// One WG owns 64 rows for all 20 timesteps. LDS: fused weights (256 KB) +
// double-buffered h-state (2 x 16 KB) = 294912 B of the WGP's 320 KB.
// c-state lives in VGPRs for all 20 steps. gates = [x|h] @ Wc^T + bsum.
// ---------------------------------------------------------------------------
__global__ void lstm_kernel(const int* __restrict__ agg,
                            const unsigned short* __restrict__ hpB,
                            const unsigned short* __restrict__ WcBt,
                            const float* __restrict__ bsum,
                            unsigned short* __restrict__ hnB,
                            float* __restrict__ hnF, float* __restrict__ cnF) {
  extern __shared__ __align__(16) unsigned short smem[];
  unsigned short* Wc    = smem;               // 512*256 bf16
  unsigned short* hbuf0 = smem + 512 * 256;   // 64*128 bf16
  unsigned short* hbuf1 = hbuf0 + 64 * 128;   // 64*128 bf16
  const int tid = threadIdx.x;
  {
    uint4* d = (uint4*)Wc;
    const uint4* s = (const uint4*)WcBt;
    for (int i = tid; i < (512 * 256) / 8; i += blockDim.x) d[i] = s[i];
    uint4* z = (uint4*)hbuf0;
    for (int i = tid; i < (64 * 128) / 8; i += blockDim.x) z[i] = make_uint4(0, 0, 0, 0);
  }
  __syncthreads();

  const int wave = tid >> 5, lane = tid & 31;
  const int ln = lane & 15, hi = lane >> 4;
  const int row0 = blockIdx.x * 64;

  float bias[4];
#pragma unroll
  for (int g = 0; g < 4; ++g) bias[g] = bsum[g * 128 + wave * 16 + ln];

  v8f cstate[4];
#pragma unroll
  for (int rt = 0; rt < 4; ++rt) cstate[rt] = (v8f){};

  for (int t = 0; t < KNEI; ++t) {
    unsigned short* hrd = (t & 1) ? hbuf1 : hbuf0;
    unsigned short* hwr = (t & 1) ? hbuf0 : hbuf1;

    for (int rt = 0; rt < 4; ++rt) {
      int row = row0 + rt * 16 + ln;
      int rowc = row < N_NODES ? row : N_NODES - 1;
      int src = agg[rowc * KNEI + t];
      const unsigned short* xrow = hpB + (size_t)src * DDIM;         // gathered x_t row
      const unsigned short* hrow = hrd + (rt * 16 + ln) * DDIM;      // h_{t-1} row (LDS)

      v8f acc[4];
#pragma unroll
      for (int g = 0; g < 4; ++g) {
        v8f z;
#pragma unroll
        for (int v = 0; v < 8; ++v) z[v] = bias[g];
        acc[g] = z;
      }

#pragma unroll
      for (int kc = 0; kc < 8; ++kc) {
        FragBF a;
        if (kc < 4) {                                   // K in [0,128): x_t (global gather)
          int k0 = kc * 32;
          a.q[0] = *(const uint4*)(xrow + k0 + 8 * hi);
          a.q[1] = *(const uint4*)(xrow + k0 + 16 + 8 * hi);
        } else {                                        // K in [128,256): h_{t-1} (LDS)
          int k0 = (kc - 4) * 32;
          a.q[0] = *(const uint4*)(hrow + k0 + 8 * hi);
          a.q[1] = *(const uint4*)(hrow + k0 + 16 + 8 * hi);
        }
        const int k0c = kc * 32;
#pragma unroll
        for (int g = 0; g < 4; ++g) {
          const unsigned short* brow = Wc + (g * 128 + wave * 16 + ln) * KC;
          FragBF b;
          b.q[0] = *(const uint4*)(brow + k0c + 8 * hi);
          b.q[1] = *(const uint4*)(brow + k0c + 16 + 8 * hi);
          acc[g] = __builtin_amdgcn_wmma_f32_16x16x32_bf16(false, a.v, false, b.v,
                                                           (short)0, acc[g], false, false);
        }
      }

      // pointwise LSTM cell, entirely in registers
      v8f cn, hn;
#pragma unroll
      for (int v = 0; v < 8; ++v) {
        float iv = sigmoidf_(acc[0][v]);
        float fv = sigmoidf_(acc[1][v]);
        float gv = tanhf(acc[2][v]);
        float ov = sigmoidf_(acc[3][v]);
        float c  = fv * cstate[rt][v] + iv * gv;
        cn[v] = c;
        hn[v] = ov * tanhf(c);
      }
      cstate[rt] = cn;

      const int col = wave * 16 + ln;
#pragma unroll
      for (int v = 0; v < 8; ++v) {
        int m = v + 8 * hi;                              // C layout row
        hwr[(rt * 16 + m) * DDIM + col] = f2bf(hn[v]);
      }
      if (t == KNEI - 1) {
#pragma unroll
        for (int v = 0; v < 8; ++v) {
          int r = row0 + rt * 16 + v + 8 * hi;
          if (r < N_NODES) {
            hnF[(size_t)r * DDIM + col] = hn[v];
            cnF[(size_t)r * DDIM + col] = cn[v];
            hnB[(size_t)r * DDIM + col] = f2bf(hn[v]);
          }
        }
      }
    }
    __syncthreads();   // step t writes visible before step t+1 reads
  }
}

// ---------------------------------------------------------------------------
// Kernel 3: out = elu([hp | h_n] @ W)    (N x 256)@(256 x 128)
// ---------------------------------------------------------------------------
__global__ void out_kernel(const unsigned short* __restrict__ hpB,
                           const unsigned short* __restrict__ hnB,
                           const unsigned short* __restrict__ WBt,
                           float* __restrict__ out) {
  extern __shared__ __align__(16) unsigned short smem[];   // 64 KB: WBt
  const int tid = threadIdx.x;
  {
    uint4* d = (uint4*)smem;
    const uint4* s = (const uint4*)WBt;
    for (int i = tid; i < (128 * 256) / 8; i += blockDim.x) d[i] = s[i];
  }
  __syncthreads();
  const int wave = tid >> 5, lane = tid & 31;
  const int ln = lane & 15, hi = lane >> 4;
  const int row0 = blockIdx.x * 64;

  for (int rt = 0; rt < 4; ++rt) {
    int row = row0 + rt * 16 + ln;
    int rowc = row < N_NODES ? row : N_NODES - 1;
    const unsigned short* prow = hpB + (size_t)rowc * DDIM;
    const unsigned short* nrow = hnB + (size_t)rowc * DDIM;
    v8f acc = {};
#pragma unroll
    for (int kc = 0; kc < 8; ++kc) {
      FragBF a;
      if (kc < 4) {
        int k0 = kc * 32;
        a.q[0] = *(const uint4*)(prow + k0 + 8 * hi);
        a.q[1] = *(const uint4*)(prow + k0 + 16 + 8 * hi);
      } else {
        int k0 = (kc - 4) * 32;
        a.q[0] = *(const uint4*)(nrow + k0 + 8 * hi);
        a.q[1] = *(const uint4*)(nrow + k0 + 16 + 8 * hi);
      }
      const unsigned short* brow = smem + (wave * 16 + ln) * KC;
      FragBF b;
      int k0c = kc * 32;
      b.q[0] = *(const uint4*)(brow + k0c + 8 * hi);
      b.q[1] = *(const uint4*)(brow + k0c + 16 + 8 * hi);
      acc = __builtin_amdgcn_wmma_f32_16x16x32_bf16(false, a.v, false, b.v,
                                                    (short)0, acc, false, false);
    }
    const int col = wave * 16 + ln;
#pragma unroll
    for (int v = 0; v < 8; ++v) {
      int r = row0 + rt * 16 + v + 8 * hi;
      if (r < N_NODES) {
        float x = acc[v];
        out[(size_t)r * DDIM + col] = x > 0.f ? x : (expf(x) - 1.0f);
      }
    }
  }
}

// ---------------------------------------------------------------------------
extern "C" void kernel_launch(void* const* d_in, const int* in_sizes, int n_in,
                              void* d_out, int out_size, void* d_ws, size_t ws_size,
                              hipStream_t stream) {
  (void)in_sizes; (void)n_in; (void)out_size; (void)ws_size;
  const float* h     = (const float*)d_in[0];
  const int*   agg   = (const int*)d_in[1];
  const float* Wpast = (const float*)d_in[2];
  const float* W     = (const float*)d_in[3];
  const float* Wih   = (const float*)d_in[4];
  const float* Whh   = (const float*)d_in[5];
  const float* bih   = (const float*)d_in[6];
  const float* bhh   = (const float*)d_in[7];

  // workspace layout (bytes), all offsets 16B aligned
  char* ws = (char*)d_ws;
  unsigned short* hpB  = (unsigned short*)(ws);                 // N*128 bf16 = 12.8 MB
  unsigned short* hnB  = (unsigned short*)(ws + 12800000);      // N*128 bf16 = 12.8 MB
  unsigned short* hB   = (unsigned short*)(ws + 25600000);      // N*256 bf16 = 25.6 MB
  unsigned short* WpBt = (unsigned short*)(ws + 51200000);      // 64 KB
  unsigned short* WBt  = (unsigned short*)(ws + 51265536);      // 64 KB
  unsigned short* WcBt = (unsigned short*)(ws + 51331072);      // 256 KB
  float*          bsum = (float*)(ws + 51593216);               // 2 KB

  float* out = (float*)d_out;                                   // (N,128)
  float* hnF = out + (size_t)N_NODES * DDIM;                    // (N,128)
  float* cnF = out + 2 * (size_t)N_NODES * DDIM;                // (N,128)

  // allow >64KB dynamic LDS (CDNA5 WGP has 320 KB); idempotent, capture-safe
  (void)hipFuncSetAttribute((const void*)lstm_kernel,
                            hipFuncAttributeMaxDynamicSharedMemorySize, 294912);
  (void)hipFuncSetAttribute((const void*)hp_kernel,
                            hipFuncAttributeMaxDynamicSharedMemorySize, 65536);
  (void)hipFuncSetAttribute((const void*)out_kernel,
                            hipFuncAttributeMaxDynamicSharedMemorySize, 65536);

  const int WGS = (N_NODES + 63) / 64;   // 782 workgroups of 64 rows

  prep_kernel<<<512, 256, 0, stream>>>(Wpast, W, Wih, Whh, bih, bhh,
                                       WpBt, WBt, WcBt, bsum);
  hcvt_kernel<<<(N_NODES * FDIM / 8 + 255) / 256, 256, 0, stream>>>(h, hB);
  hp_kernel<<<WGS, 256, 65536, stream>>>(hB, hpB, WpBt);
  lstm_kernel<<<WGS, 256, 294912, stream>>>(agg, hpB, WcBt, bsum, hnB, hnF, cnF);
  out_kernel<<<WGS, 256, 65536, stream>>>(hpB, hnB, WBt, out);
}
